// MultiLabelEvolveGCN_78228534329935
// MI455X (gfx1250) — compile-verified, matrix operand order checked by reading, not run on hardware
//
#include <hip/hip_runtime.h>
#include <math.h>

// ---------------- types ----------------
typedef __attribute__((ext_vector_type(16))) __bf16          v16bf;
typedef __attribute__((ext_vector_type(16))) unsigned short  v16us;
typedef __attribute__((ext_vector_type(8)))  float           v8f;

static constexpr int N_NODES = 100000;
static constexpr int D_DIM   = 128;
static constexpr int L_DIM   = 100;
static constexpr int E_EDGES = 625000;

// ---------------- helpers ----------------
__device__ __forceinline__ unsigned short f2bf(float f) {
  unsigned int u = __builtin_bit_cast(unsigned int, f);
  unsigned int r = (u + 0x7FFFu + ((u >> 16) & 1u)) >> 16;   // round-to-nearest-even
  return (unsigned short)r;
}
__device__ __forceinline__ unsigned int f2k(float f) {       // monotonic float->u32 key
  unsigned int u = __builtin_bit_cast(unsigned int, f);
  return (u & 0x80000000u) ? ~u : (u | 0x80000000u);
}
__device__ __forceinline__ float k2f(unsigned int k) {       // inverse of f2k
  unsigned int u = (k & 0x80000000u) ? (k & 0x7FFFFFFFu) : ~k;
  return __builtin_bit_cast(float, u);
}
__device__ __forceinline__ v8f zero8() {
  v8f z;
#pragma unroll
  for (int i = 0; i < 8; ++i) z[i] = 0.0f;
  return z;
}
// Offset (in shorts) of element (krel, lr) of fragment (ks, nt) in a
// fragment-major LDS buffer: each fragment = 32 lanes * 16 contiguous bf16.
__device__ __forceinline__ int bfrag_off(int ks, int nt, int krel, int lr, int NT) {
  int half = (krel >> 3) & 1;
  int e    = (krel & 7) | ((krel & 16) >> 1);
  int lane = half * 16 + lr;
  return (((ks * NT + nt) * 32) + lane) * 16 + e;
}

// state layout (u32 words): [0]=prefix [1]=k_remaining [2]=cnt_gt [3]=cnt_eq
//                           [4]=inv_pnorm(float bits)  [16..271]=hist[256]
static constexpr int STATE_WORDS = 272;

// ---------------- stage 0: init / projection score ----------------
__global__ void k_init(unsigned int* state) {
  for (int i = threadIdx.x; i < STATE_WORDS; i += blockDim.x)
    state[i] = (i == 1) ? 128u : 0u;
}

__global__ void pnorm_kernel(const float* __restrict__ p, unsigned int* state) {
  int l = threadIdx.x;                       // 32 threads
  float s = 0.0f;
  for (int k = l; k < D_DIM; k += 32) { float v = p[k]; s += v * v; }
#pragma unroll
  for (int off = 16; off > 0; off >>= 1) s += __shfl_down(s, off, 32);
  if (l == 0) ((float*)state)[4] = 1.0f / sqrtf(s);
}

__global__ void score_kernel(const float* __restrict__ x, const float* __restrict__ p,
                             const unsigned int* __restrict__ state,
                             float* __restrict__ score, int n) {
  int wave = threadIdx.x >> 5, lane = threadIdx.x & 31;
  int r = blockIdx.x * 8 + wave;
  if (r >= n) return;
  float inv = ((const float*)state)[4];
  float s = 0.0f;
#pragma unroll
  for (int q = 0; q < 4; ++q) { int k = lane + q * 32; s += x[(size_t)r * D_DIM + k] * p[k]; }
#pragma unroll
  for (int off = 16; off > 0; off >>= 1) s += __shfl_down(s, off, 32);
  if (lane == 0) score[r] = s * inv;
}

// ---------------- stage 1: deterministic radix-select top-128 ----------------
__global__ void hist_kernel(const float* __restrict__ score, unsigned int* state,
                            int round, int n) {
  __shared__ unsigned int lh[256];
  unsigned int* hist = state + 16;
  lh[threadIdx.x] = 0;   // blockDim == 256
  __syncthreads();
  unsigned int prefix = state[0];
  int dshift = 8 * (3 - round);
  for (int i = blockIdx.x * blockDim.x + threadIdx.x; i < n; i += gridDim.x * blockDim.x) {
    unsigned int key = f2k(score[i]);
    bool match = (round == 0) || ((key >> (8 * (4 - round))) == prefix);
    if (match) atomicAdd(&lh[(key >> dshift) & 255u], 1u);
  }
  __syncthreads();
  if (lh[threadIdx.x]) atomicAdd(&hist[threadIdx.x], lh[threadIdx.x]);
}

__global__ void scan_kernel(unsigned int* state) {
  unsigned int* hist = state + 16;
  unsigned int krem = state[1];
  unsigned int cum = 0; int chosen = 0;
  for (int d = 255; d >= 0; --d) {
    unsigned int c = hist[d];
    if (cum + c >= krem) { chosen = d; break; }
    cum += c;
  }
  state[0] = (state[0] << 8) | (unsigned int)chosen;
  state[1] = krem - cum;
  for (int d = 0; d < 256; ++d) hist[d] = 0;
}

__global__ void collect_kernel(const float* __restrict__ score, unsigned int* state,
                               unsigned int* sel_key, int* sel_idx, int* eq_idx, int n) {
  unsigned int T = state[0];
  for (int i = blockIdx.x * blockDim.x + threadIdx.x; i < n; i += gridDim.x * blockDim.x) {
    unsigned int key = f2k(score[i]);
    if (key > T) {
      unsigned int pos = atomicAdd(&state[2], 1u);
      if (pos < 128u) { sel_key[pos] = key; sel_idx[pos] = i; }
    } else if (key == T) {
      unsigned int pos = atomicAdd(&state[3], 1u);
      if (pos < 256u) eq_idx[pos] = i;
    }
  }
}

__global__ void assemble_kernel(unsigned int* state, const unsigned int* sel_key,
                                const int* sel_idx, int* eq_idx,
                                int* top_idx, float* gate) {
  __shared__ unsigned int sk[128];
  __shared__ int si[128];
  int t = threadIdx.x;                              // 128 threads
  unsigned int g = state[2]; if (g > 128u) g = 128u;
  unsigned int T = state[0];
  if ((unsigned)t < g) { sk[t] = sel_key[t]; si[t] = sel_idx[t]; }
  __syncthreads();
  if (t == 0) {
    int ec = (int)state[3]; if (ec > 256) ec = 256;
    for (int a = 1; a < ec; ++a) {                  // sort ties by index ascending
      int v = eq_idx[a]; int b = a - 1;
      while (b >= 0 && eq_idx[b] > v) { eq_idx[b + 1] = eq_idx[b]; --b; }
      eq_idx[b + 1] = v;
    }
    int need = 128 - (int)g;
    for (int j = 0; j < need; ++j) { sk[g + j] = T; si[g + j] = (j < ec) ? eq_idx[j] : 0; }
  }
  __syncthreads();
  // bitonic sort: key descending, index ascending on ties (matches lax.top_k)
  for (int kk = 2; kk <= 128; kk <<= 1) {
    for (int jj = kk >> 1; jj > 0; jj >>= 1) {
      int ixj = t ^ jj;
      if (ixj > t) {
        unsigned int k1 = sk[t], k2 = sk[ixj];
        int i1 = si[t], i2 = si[ixj];
        bool lt = (k1 > k2) || (k1 == k2 && i1 < i2);   // "t before ixj" in desired order
        bool up = ((t & kk) == 0);
        if (up ? !lt : lt) { sk[t] = k2; sk[ixj] = k1; si[t] = i2; si[ixj] = i1; }
      }
      __syncthreads();
    }
  }
  top_idx[t] = si[t];
  gate[t] = tanhf(k2f(sk[t]));
}

// ---------------- stage 2: gather + GRU weight evolution (fp32) ----------------
__global__ void xtilde_kernel(const float* __restrict__ x, const int* __restrict__ top_idx,
                              const float* __restrict__ gate, float* __restrict__ xt) {
  int r = blockIdx.x, c = threadIdx.x;
  xt[r * D_DIM + c] = x[(size_t)top_idx[r] * D_DIM + c] * gate[r];
}

__global__ void gru_kernel(const float* __restrict__ xt, const float* __restrict__ w0,
                           const float* __restrict__ w_ih, const float* __restrict__ w_hh,
                           const float* __restrict__ b_ih, const float* __restrict__ b_hh,
                           float* __restrict__ Wm) {
  __shared__ float xr[D_DIM], hrow[D_DIM];
  int r = blockIdx.x, c = threadIdx.x;
  xr[c] = xt[r * D_DIM + c];
  hrow[c] = w0[r * D_DIM + c];
  __syncthreads();
  float aI0 = b_ih[c], aI1 = b_ih[128 + c], aI2 = b_ih[256 + c];
  float aH0 = b_hh[c], aH1 = b_hh[128 + c], aH2 = b_hh[256 + c];
  for (int k = 0; k < D_DIM; ++k) {
    float xv = xr[k], hv = hrow[k];
    aI0 += xv * w_ih[(0 * 128 + c) * D_DIM + k];
    aI1 += xv * w_ih[(1 * 128 + c) * D_DIM + k];
    aI2 += xv * w_ih[(2 * 128 + c) * D_DIM + k];
    aH0 += hv * w_hh[(0 * 128 + c) * D_DIM + k];
    aH1 += hv * w_hh[(1 * 128 + c) * D_DIM + k];
    aH2 += hv * w_hh[(2 * 128 + c) * D_DIM + k];
  }
  float rg = 1.0f / (1.0f + expf(-(aI0 + aH0)));
  float zg = 1.0f / (1.0f + expf(-(aI1 + aH1)));
  float ng = tanhf(aI2 + rg * aH2);
  Wm[r * D_DIM + c] = (1.0f - zg) * ng + zg * hrow[c];
}

// ---------------- stage 3: xw = x @ W  via v_wmma_f32_16x16x32_bf16 ----------
// W is staged in LDS pre-swizzled into WMMA B-fragment layout so the inner
// loop loads each fragment with wide contiguous LDS reads (2 x b128 / lane).
__global__ void __launch_bounds__(256)
gemm_xw_kernel(const float* __restrict__ x, const float* __restrict__ Wm,
               float* __restrict__ xw, int n) {
  __shared__ unsigned short Wl[4 * 8 * 512];     // 32 fragments * 512 bf16 = 32KB
  int tid = threadIdx.x;
  // stage W: 4 consecutive K per thread-iteration -> one ushort4 LDS store
  for (int i = tid; i < (128 * 128) / 4; i += 256) {
    int kq = i >> 7;            // 0..31
    int nn = i & 127;
    int k  = kq << 2;
    ushort4 v;
    v.x = f2bf(Wm[(k + 0) * D_DIM + nn]);
    v.y = f2bf(Wm[(k + 1) * D_DIM + nn]);
    v.z = f2bf(Wm[(k + 2) * D_DIM + nn]);
    v.w = f2bf(Wm[(k + 3) * D_DIM + nn]);
    int off = bfrag_off(k >> 5, nn >> 4, k & 31, nn & 15, 8);  // e in {0,4,8,12}
    *reinterpret_cast<ushort4*>(&Wl[off]) = v;
  }
  __syncthreads();

  int wave = tid >> 5, lane = tid & 31;
  int half = lane >> 4;                          // 0: K 0..7/16..23, 1: K 8..15/24..31
  int lr   = lane & 15;
  int r0   = blockIdx.x * 128;
  int m0   = wave * 16;
  int gm   = r0 + m0 + lr;                       // A-fragment row for this lane
  const float* xrow = x + (size_t)((gm < n) ? gm : 0) * D_DIM;

  v8f acc[8];
#pragma unroll
  for (int t = 0; t < 8; ++t) acc[t] = zero8();

#pragma unroll
  for (int ks = 0; ks < 4; ++ks) {
    int k0 = ks * 32 + half * 8;
    const float4* vp = reinterpret_cast<const float4*>(xrow + k0);
    float4 A0 = vp[0], A1 = vp[1], A2 = vp[4], A3 = vp[5];
    v16us au;
    au[0]=f2bf(A0.x);  au[1]=f2bf(A0.y);  au[2]=f2bf(A0.z);  au[3]=f2bf(A0.w);
    au[4]=f2bf(A1.x);  au[5]=f2bf(A1.y);  au[6]=f2bf(A1.z);  au[7]=f2bf(A1.w);
    au[8]=f2bf(A2.x);  au[9]=f2bf(A2.y);  au[10]=f2bf(A2.z); au[11]=f2bf(A2.w);
    au[12]=f2bf(A3.x); au[13]=f2bf(A3.y); au[14]=f2bf(A3.z); au[15]=f2bf(A3.w);
    v16bf a = __builtin_bit_cast(v16bf, au);
#pragma unroll
    for (int nt = 0; nt < 8; ++nt) {
      v16us bu = *reinterpret_cast<const v16us*>(
          &Wl[(((ks * 8 + nt) * 32) + lane) * 16]);   // 32B contiguous per lane
      v16bf b = __builtin_bit_cast(v16bf, bu);
      acc[nt] = __builtin_amdgcn_wmma_f32_16x16x32_bf16(
          false, a, false, b, (short)0, acc[nt], false, false);
    }
  }
#pragma unroll
  for (int nt = 0; nt < 8; ++nt) {
#pragma unroll
    for (int i = 0; i < 8; ++i) {
      int m = r0 + m0 + half * 8 + i;            // D layout: m = 8*half + i, n = lr
      if (m < n) xw[(size_t)m * D_DIM + nt * 16 + lr] = acc[nt][i];
    }
  }
}

// ---------------- stage 4: degrees / norm / scatter (memory-bound) ----------
__global__ void deg_init_kernel(float* deg, int n) {
  int i = blockIdx.x * blockDim.x + threadIdx.x;
  if (i < n) deg[i] = 1.0f;                      // self-loop contributes 1
}
__global__ void deg_count_kernel(const int* __restrict__ cols, float* deg, int e) {
  for (int i = blockIdx.x * blockDim.x + threadIdx.x; i < e; i += gridDim.x * blockDim.x)
    atomicAdd(&deg[cols[i]], 1.0f);
}
__global__ void dis_kernel(float* deg, int n) {
  int i = blockIdx.x * blockDim.x + threadIdx.x;
  if (i < n) deg[i] = rsqrtf(deg[i]);            // deg >= 1 always
}
__global__ void inith_kernel(const float* __restrict__ dis, const float* __restrict__ xw,
                             float* __restrict__ h, int n) {
  int r = blockIdx.x * 2 + (threadIdx.x >> 7);
  int c = threadIdx.x & 127;
  if (r < n) { float d = dis[r]; h[(size_t)r * D_DIM + c] = d * d * xw[(size_t)r * D_DIM + c]; }
}
__global__ void scatter_kernel(const int* __restrict__ rows, const int* __restrict__ cols,
                               const float* __restrict__ dis, const float* __restrict__ xw,
                               float* __restrict__ h, int e) {
  int idx = blockIdx.x * 2 + (threadIdx.x >> 7);
  int c = threadIdx.x & 127;
  if (idx < e) {
    int r = rows[idx], cc = cols[idx];
    float nrm = dis[r] * dis[cc];
    atomicAdd(&h[(size_t)cc * D_DIM + c], nrm * xw[(size_t)r * D_DIM + c]);
  }
}

// ---------------- stage 5: out = relu(h) @ lin_w^T + lin_b (WMMA) -----------
__global__ void __launch_bounds__(256)
gemm_out_kernel(const float* __restrict__ h, const float* __restrict__ lw,
                const float* __restrict__ lb, float* __restrict__ out, int n) {
  constexpr int LPAD = 112;                      // 100 cols padded to 7 tiles of 16
  __shared__ unsigned short Bl[4 * 7 * 512];     // 28 fragments * 512 bf16 = 28KB
  int tid = threadIdx.x;
  // stage lin_w^T pre-swizzled: 4 consecutive K per iteration
  for (int i = tid; i < (128 * LPAD) / 4; i += 256) {
    int kq = i / LPAD;          // 0..31
    int l  = i - kq * LPAD;     // 0..111
    int k  = kq << 2;
    ushort4 v;
    v.x = (l < L_DIM) ? f2bf(lw[l * D_DIM + k + 0]) : (unsigned short)0;
    v.y = (l < L_DIM) ? f2bf(lw[l * D_DIM + k + 1]) : (unsigned short)0;
    v.z = (l < L_DIM) ? f2bf(lw[l * D_DIM + k + 2]) : (unsigned short)0;
    v.w = (l < L_DIM) ? f2bf(lw[l * D_DIM + k + 3]) : (unsigned short)0;
    int off = bfrag_off(k >> 5, l >> 4, k & 31, l & 15, 7);
    *reinterpret_cast<ushort4*>(&Bl[off]) = v;
  }
  __syncthreads();

  int wave = tid >> 5, lane = tid & 31;
  int half = lane >> 4;
  int lr   = lane & 15;
  int r0   = blockIdx.x * 128;
  int m0   = wave * 16;
  int gm   = r0 + m0 + lr;
  const float* hrow = h + (size_t)((gm < n) ? gm : 0) * D_DIM;

  v8f acc[7];
#pragma unroll
  for (int t = 0; t < 7; ++t) acc[t] = zero8();

#pragma unroll
  for (int ks = 0; ks < 4; ++ks) {
    int k0 = ks * 32 + half * 8;
    const float4* vp = reinterpret_cast<const float4*>(hrow + k0);
    float4 A0 = vp[0], A1 = vp[1], A2 = vp[4], A3 = vp[5];
    v16us au;
    au[0]=f2bf(fmaxf(A0.x,0.f));  au[1]=f2bf(fmaxf(A0.y,0.f));
    au[2]=f2bf(fmaxf(A0.z,0.f));  au[3]=f2bf(fmaxf(A0.w,0.f));
    au[4]=f2bf(fmaxf(A1.x,0.f));  au[5]=f2bf(fmaxf(A1.y,0.f));
    au[6]=f2bf(fmaxf(A1.z,0.f));  au[7]=f2bf(fmaxf(A1.w,0.f));
    au[8]=f2bf(fmaxf(A2.x,0.f));  au[9]=f2bf(fmaxf(A2.y,0.f));
    au[10]=f2bf(fmaxf(A2.z,0.f)); au[11]=f2bf(fmaxf(A2.w,0.f));
    au[12]=f2bf(fmaxf(A3.x,0.f)); au[13]=f2bf(fmaxf(A3.y,0.f));
    au[14]=f2bf(fmaxf(A3.z,0.f)); au[15]=f2bf(fmaxf(A3.w,0.f));
    v16bf a = __builtin_bit_cast(v16bf, au);
#pragma unroll
    for (int nt = 0; nt < 7; ++nt) {
      v16us bu = *reinterpret_cast<const v16us*>(
          &Bl[(((ks * 7 + nt) * 32) + lane) * 16]);
      v16bf b = __builtin_bit_cast(v16bf, bu);
      acc[nt] = __builtin_amdgcn_wmma_f32_16x16x32_bf16(
          false, a, false, b, (short)0, acc[nt], false, false);
    }
  }
#pragma unroll
  for (int nt = 0; nt < 7; ++nt) {
    int col = nt * 16 + lr;
    float bias = (col < L_DIM) ? lb[col] : 0.0f;
#pragma unroll
    for (int i = 0; i < 8; ++i) {
      int m = r0 + m0 + half * 8 + i;
      if (m < n && col < L_DIM) out[(size_t)m * L_DIM + col] = acc[nt][i] + bias;
    }
  }
}

// ---------------- launcher ----------------
extern "C" void kernel_launch(void* const* d_in, const int* in_sizes, int n_in,
                              void* d_out, int out_size, void* d_ws, size_t ws_size,
                              hipStream_t stream) {
  (void)in_sizes; (void)n_in; (void)out_size; (void)ws_size;
  const float* x     = (const float*)d_in[0];
  const int*   ei    = (const int*)  d_in[1];
  const float* w0    = (const float*)d_in[2];
  const float* p     = (const float*)d_in[3];
  const float* w_ih  = (const float*)d_in[4];
  const float* w_hh  = (const float*)d_in[5];
  const float* b_ih  = (const float*)d_in[6];
  const float* b_hh  = (const float*)d_in[7];
  const float* lin_w = (const float*)d_in[8];
  const float* lin_b = (const float*)d_in[9];
  float* out = (float*)d_out;

  char* ws = (char*)d_ws;
  size_t off = 0;
  auto take = [&](size_t bytes) -> char* {
    char* ptr = ws + off;
    off = (off + bytes + 511) & ~(size_t)511;
    return ptr;
  };
  float*        score   = (float*)take((size_t)N_NODES * 4);
  unsigned int* state   = (unsigned int*)take(STATE_WORDS * 4);
  unsigned int* sel_key = (unsigned int*)take(128 * 4);
  int*          sel_idx = (int*)take(128 * 4);
  int*          eq_idx  = (int*)take(256 * 4);
  int*          top_idx = (int*)take(128 * 4);
  float*        gate    = (float*)take(128 * 4);
  float*        xt      = (float*)take((size_t)D_DIM * D_DIM * 4);
  float*        Wm      = (float*)take((size_t)D_DIM * D_DIM * 4);
  float*        dis     = (float*)take((size_t)N_NODES * 4);
  float*        xw      = (float*)take((size_t)N_NODES * D_DIM * 4);
  float*        h       = (float*)take((size_t)N_NODES * D_DIM * 4);

  const int* erows = ei;             // edge_index[0]  (sources)
  const int* ecols = ei + E_EDGES;   // edge_index[1]  (targets)

  // top-k selection + gating
  k_init<<<1, 256, 0, stream>>>(state);
  pnorm_kernel<<<1, 32, 0, stream>>>(p, state);
  score_kernel<<<(N_NODES + 7) / 8, 256, 0, stream>>>(x, p, state, score, N_NODES);
  for (int r = 0; r < 4; ++r) {
    hist_kernel<<<512, 256, 0, stream>>>(score, state, r, N_NODES);
    scan_kernel<<<1, 1, 0, stream>>>(state);
  }
  collect_kernel<<<512, 256, 0, stream>>>(score, state, sel_key, sel_idx, eq_idx, N_NODES);
  assemble_kernel<<<1, 128, 0, stream>>>(state, sel_key, sel_idx, eq_idx, top_idx, gate);

  // GRU weight evolution (fp32)
  xtilde_kernel<<<128, 128, 0, stream>>>(x, top_idx, gate, xt);
  gru_kernel<<<128, 128, 0, stream>>>(xt, w0, w_ih, w_hh, b_ih, b_hh, Wm);

  // xw = x @ W   (WMMA bf16 -> f32)
  gemm_xw_kernel<<<(N_NODES + 127) / 128, 256, 0, stream>>>(x, Wm, xw, N_NODES);

  // symmetric-norm aggregation
  deg_init_kernel<<<(N_NODES + 255) / 256, 256, 0, stream>>>(dis, N_NODES);
  deg_count_kernel<<<512, 256, 0, stream>>>(ecols, dis, E_EDGES);
  dis_kernel<<<(N_NODES + 255) / 256, 256, 0, stream>>>(dis, N_NODES);
  inith_kernel<<<(N_NODES + 1) / 2, 256, 0, stream>>>(dis, xw, h, N_NODES);
  scatter_kernel<<<(E_EDGES + 1) / 2, 256, 0, stream>>>(erows, ecols, dis, xw, h, E_EDGES);

  // out = relu(h) @ lin_w^T + lin_b   (WMMA bf16 -> f32)
  gemm_out_kernel<<<(N_NODES + 127) / 128, 256, 0, stream>>>(h, lin_w, lin_b, out, N_NODES);
}
